// BasicLSTM_49967649522039
// MI455X (gfx1250) — matrix operand order repaired and checked by reading, not compiled, not run on hardware
//
#include <hip/hip_runtime.h>

typedef _Float16 half_t;
typedef __attribute__((ext_vector_type(16))) _Float16 v16h;
typedef __attribute__((ext_vector_type(8)))  _Float16 v8h;
typedef __attribute__((ext_vector_type(4)))  _Float16 v4h;
typedef __attribute__((ext_vector_type(8)))  float    v8f;
typedef __attribute__((ext_vector_type(4)))  float    v4f;

#define B_    64
#define T_    12
#define N_    1370
#define C_    8
#define H_    64
#define G4H   256       // 4*H gate columns
#define KDIM  96        // K: 64 (h) + 8 (x) + 1 (bias one-hot) + pad -> 3 chunks of 32
#define KBIAS 72        // constant-1 column carrying the bias row of Bt
#define WAVES 2
#define ROWS  16        // sequences (M rows) per wave

// ---- fast element-wise math on gfx1250 hardware trans units ----
__device__ __forceinline__ float fast_tanh(float x) {
#if __has_builtin(__builtin_amdgcn_tanhf)
    return __builtin_amdgcn_tanhf(x);          // v_tanh_f32
#else
    // 2*sigmoid(2x)-1 via v_exp_f32 / v_rcp_f32
    float e = __builtin_amdgcn_exp2f(x * -2.8853900817779268f);  // exp(-2x)
    return 2.0f * __builtin_amdgcn_rcpf(1.0f + e) - 1.0f;
#endif
}
__device__ __forceinline__ float fast_sigmoid(float x) {
    float e = __builtin_amdgcn_exp2f(x * -1.4426950408889634f);  // exp(-x)
    return __builtin_amdgcn_rcpf(1.0f + e);
}

__global__ __launch_bounds__(WAVES * 32)
void lstm_wmma_kernel(const float* __restrict__ x,
                      const float* __restrict__ W_ih,
                      const float* __restrict__ W_hh,
                      const float* __restrict__ b_ih,
                      const float* __restrict__ b_hh,
                      const float* __restrict__ W_fc,
                      const float* __restrict__ b_fc,
                      float* __restrict__ out)
{
    // Combined weights, K-contiguous per gate column:
    // Bt[g][k] = k<64: W_hh[g][k] ; 64..71: W_ih[g][k-64] ; 72: b_ih[g]+b_hh[g] ; else 0
    __shared__ half_t Bt[G4H][KDIM];
    __shared__ float  Wfc[C_][H_];
    __shared__ float  bfc[C_];
    // Per-wave A staging: [row][k]; k<64 = h(f16), 64..71 = x_t, 72 = 1.0, rest 0
    __shared__ half_t Abuf[WAVES][ROWS][KDIM];
    // Per-wave raw f32 x staging, double buffered, filled by async-to-LDS DMA
    __shared__ float  xstage[WAVES][2][ROWS][C_];

    const int tid  = threadIdx.x;
    const int wave = tid >> 5;
    const int lane = tid & 31;

    // ---- cooperative weight staging ----
    for (int idx = tid; idx < G4H * KDIM; idx += WAVES * 32) {
        int g = idx / KDIM, k = idx - g * KDIM;
        float w = 0.f;
        if (k < H_)            w = W_hh[g * H_ + k];
        else if (k < H_ + C_)  w = W_ih[g * C_ + (k - H_)];
        else if (k == KBIAS)   w = b_ih[g] + b_hh[g];
        Bt[g][k] = (half_t)w;
    }
    for (int idx = tid; idx < C_ * H_; idx += WAVES * 32)
        Wfc[idx / H_][idx % H_] = W_fc[idx];
    if (tid < C_) bfc[tid] = b_fc[tid];

    // zero this wave's A buffer (h0 = 0, pad = 0), then set bias one-hot column
    {
        v8h zv = {0, 0, 0, 0, 0, 0, 0, 0};
        v8h* ab = (v8h*)&Abuf[wave][0][0];
        for (int idx = lane; idx < ROWS * KDIM / 8; idx += 32) ab[idx] = zv;
        if (lane < ROWS) Abuf[wave][lane][KBIAS] = (half_t)1.0f;
    }
    __syncthreads();

    // ---- per-lane x / output addressing: lane -> (row = lane/2, 4 cols) ----
    const int xrow  = lane >> 1;
    const int xcol0 = (lane & 1) * 4;
    const int s_base = (blockIdx.x * WAVES + wave) * ROWS;
    const int s  = s_base + xrow;            // global sequence id in [0, B*N)
    const int bb = s / N_;
    const int nn = s - bb * N_;
    const float* xptr = x + ((size_t)(bb * T_) * N_ + nn) * C_ + xcol0;  // t = 0
    const size_t xstep = (size_t)N_ * C_;

    const unsigned lds_x0 = (unsigned)(uintptr_t)&xstage[wave][0][xrow][xcol0];
    const unsigned lds_x1 = (unsigned)(uintptr_t)&xstage[wave][1][xrow][xcol0];

    // prologue: kick async DMA of x_0 into LDS (16B per lane)
    asm volatile("global_load_async_to_lds_b128 %0, %1, off"
                 :: "v"(lds_x0), "v"((unsigned long long)(uintptr_t)xptr) : "memory");
    xptr += xstep;

    // cell state in registers: 4 hidden slices x 8 elems (C/D layout)
    float cst[4][8];
#pragma unroll
    for (int j = 0; j < 4; ++j)
#pragma unroll
        for (int v = 0; v < 8; ++v) cst[j][v] = 0.f;

    const int acol   = lane & 15;            // N column within 16-wide C/D tile
    const int arowhi = (lane >> 4) * 8;      // +8 M rows for lanes 16..31
    const int arow   = lane & 15;            // A-frag row (M = lane%16)
    const int aoff   = (lane < 16) ? 0 : 8;  // A-frag K sub-offset
    const int bkoff  = (lane < 16) ? 0 : 8;  // B-frag K sub-offset

    for (int t = 0; t < T_; ++t) {
        // wait for this step's x tile, then immediately start DMA of the next
#if __has_builtin(__builtin_amdgcn_s_wait_asynccnt)
        __builtin_amdgcn_s_wait_asynccnt(0);
#else
        asm volatile("s_wait_asynccnt 0x0" ::: "memory");
#endif
        if (t + 1 < T_) {
            unsigned dst = ((t + 1) & 1) ? lds_x1 : lds_x0;
            asm volatile("global_load_async_to_lds_b128 %0, %1, off"
                         :: "v"(dst), "v"((unsigned long long)(uintptr_t)xptr) : "memory");
            xptr += xstep;
        }

        // convert staged f32 x_t -> f16 A-buffer cols 64..71
        v4f xv = *(const v4f*)&xstage[wave][t & 1][xrow][xcol0];
        v4h xh;
        xh.x = (half_t)xv.x; xh.y = (half_t)xv.y;
        xh.z = (half_t)xv.z; xh.w = (half_t)xv.w;
        *(v4h*)&Abuf[wave][xrow][H_ + xcol0] = xh;

        __syncthreads();   // order h/x stores vs A-fragment loads

        // A fragments for this step (shared by all 16 output tiles)
        v16h afrag[3];
#pragma unroll
        for (int kc = 0; kc < 3; ++kc) {
            const int k0 = kc * 32;
            v8h lo = *(const v8h*)&Abuf[wave][arow][k0 + aoff];
            v8h hi = *(const v8h*)&Abuf[wave][arow][k0 + aoff + 16];
#pragma unroll
            for (int q = 0; q < 8; ++q) { afrag[kc][q] = lo[q]; afrag[kc][8 + q] = hi[q]; }
        }

#pragma unroll
        for (int j = 0; j < 4; ++j) {        // hidden slice: H cols [16j, 16j+16)
            v8f acc[4];
#pragma unroll
            for (int G = 0; G < 4; ++G) acc[G] = (v8f){};
#pragma unroll
            for (int kc = 0; kc < 3; ++kc) {
                const int k0 = kc * 32;
                // issue all 4 gates' B-fragment loads (8x ds_load_b128) first,
                // then the 4 WMMAs: deepens the LDS pipeline to 8 outstanding
                v16h bf[4];
#pragma unroll
                for (int G = 0; G < 4; ++G) {
                    const half_t* bp = &Bt[G * H_ + j * 16 + acol][k0 + bkoff];
                    v8h lo = *(const v8h*)bp;
                    v8h hi = *(const v8h*)(bp + 16);
#pragma unroll
                    for (int q = 0; q < 8; ++q) { bf[G][q] = lo[q]; bf[G][8 + q] = hi[q]; }
                }
#pragma unroll
                for (int G = 0; G < 4; ++G)
                    acc[G] = __builtin_amdgcn_wmma_f32_16x16x32_f16(
                                 false, afrag[kc], false, bf[G], (short)0, acc[G],
                                 false, false);
            }
            // LSTM cell element-wise on HW trans units, write new h slice as f16
#pragma unroll
            for (int v = 0; v < 8; ++v) {
                float iv = fast_sigmoid(acc[0][v]);
                float fv = fast_sigmoid(acc[1][v]);
                float gv = fast_tanh(acc[2][v]);
                float ov = fast_sigmoid(acc[3][v]);
                float cn = fv * cst[j][v] + iv * gv;
                cst[j][v] = cn;
                float hv = ov * fast_tanh(cn);
                Abuf[wave][arowhi + v][j * 16 + acol] = (half_t)hv;
            }
        }
        __syncthreads();
    }

    // ---- final projection: y = h_T @ W_fc^T + b_fc  (16x64 . 64x8) ----
    float yo[4];
#pragma unroll
    for (int q = 0; q < 4; ++q) yo[q] = bfc[xcol0 + q];
    for (int k = 0; k < H_; ++k) {
        float hv = (float)Abuf[wave][xrow][k];
#pragma unroll
        for (int q = 0; q < 4; ++q) yo[q] += hv * Wfc[xcol0 + q][k];
    }
    v4f yv; yv.x = yo[0]; yv.y = yo[1]; yv.z = yo[2]; yv.w = yo[3];
    *(v4f*)&out[(size_t)s * C_ + xcol0] = yv;
}

extern "C" void kernel_launch(void* const* d_in, const int* in_sizes, int n_in,
                              void* d_out, int out_size, void* d_ws, size_t ws_size,
                              hipStream_t stream) {
    (void)in_sizes; (void)n_in; (void)d_ws; (void)ws_size; (void)out_size;
    const float* x    = (const float*)d_in[0];
    const float* W_ih = (const float*)d_in[1];
    const float* W_hh = (const float*)d_in[2];
    const float* b_ih = (const float*)d_in[3];
    const float* b_hh = (const float*)d_in[4];
    const float* W_fc = (const float*)d_in[5];
    const float* b_fc = (const float*)d_in[6];
    float* out = (float*)d_out;

    const int total_seq = B_ * N_;                       // 87680
    const int blocks = total_seq / (WAVES * ROWS);       // 2740 exactly
    lstm_wmma_kernel<<<blocks, WAVES * 32, 0, stream>>>(
        x, W_ih, W_hh, b_ih, b_hh, W_fc, b_fc, out);
}
